// TransformerModelAdpSpn_83648783056932
// MI455X (gfx1250) — compile-verified, matrix operand order checked by reading, not run on hardware
//
#include <hip/hip_runtime.h>

// ---------------- model constants ----------------
#define BB   8
#define MM   512
#define SS_  1024
#define HH   768
#define KH   12
#define DD   64
#define NL_  4
#define FF_  3072
#define VV   32000
#define TQ   (BB * MM)          // 4096
#define TKV  (BB * (SS_ + MM))  // 12288
#define WIN  1024
#define KEXT 1056               // key span for a 16-query block

typedef __attribute__((ext_vector_type(16))) _Float16 v16h;
typedef __attribute__((ext_vector_type(8)))  _Float16 v8h;
typedef __attribute__((ext_vector_type(8)))  float    v8f;

#define WMMA_F16(a, b, c) \
  __builtin_amdgcn_wmma_f32_16x16x32_f16(false, (a), false, (b), (short)0, (c), false, false)

// Per ISA 7.12.2 a 16-bit 16x32 fragment for lane L is two contiguous 8-half
// chunks at k-offsets kb and kb+16, where kb = (L&16)>>1.  Load them as two
// 128-bit loads and concatenate.
__device__ __forceinline__ v16h frag_chunks(const _Float16* base, int kb) {
  v8h c0 = *(const v8h*)(base + kb);
  v8h c1 = *(const v8h*)(base + kb + 16);
  return __builtin_shufflevector(c0, c1, 0, 1, 2, 3, 4, 5, 6, 7,
                                 8, 9, 10, 11, 12, 13, 14, 15);
}
// two chunks with independent (pre-clamped) start offsets
__device__ __forceinline__ v16h frag_chunks2(const _Float16* row, int s0, int s1) {
  v8h c0 = *(const v8h*)(row + s0);
  v8h c1 = *(const v8h*)(row + s1);
  return __builtin_shufflevector(c0, c1, 0, 1, 2, 3, 4, 5, 6, 7,
                                 8, 9, 10, 11, 12, 13, 14, 15);
}
// fragment from an LDS tile: lane row = lane&15, stride in halves
__device__ __forceinline__ v16h frag_lds16(const _Float16* tile, int stride, int lane) {
  return frag_chunks(tile + (lane & 15) * stride, (lane & 16) >> 1);
}

// ---------------- generic WMMA GEMM: C[T,N] = A[T,K] * W[K,N] (+bias)(+resid)(relu) ----------------
#define BT_M 128
#define BT_N 64
#define BT_K 32
#define LDS_S 40  // 32 + 8 halves pad (80B rows keep 16B alignment)

__global__ __launch_bounds__(256) void gemm_wmma_kernel(
    const float* __restrict__ A, const float* __restrict__ W,
    const float* __restrict__ bias, const float* __restrict__ resid,
    float* __restrict__ C, int T, int N, int Kd, int relu) {
  __shared__ _Float16 As[BT_M * LDS_S];
  __shared__ _Float16 Bs[BT_N * LDS_S];
  int tid = threadIdx.x, lane = tid & 31, wid = tid >> 5;
  int wr = wid >> 1, wc = wid & 1;
  int rowBase = blockIdx.y * BT_M, colBase = blockIdx.x * BT_N;
  int kb = (lane & 16) >> 1;

  v8f acc[2][2];
#pragma unroll
  for (int i = 0; i < 2; ++i)
#pragma unroll
    for (int j = 0; j < 2; ++j) acc[i][j] = (v8f)0.0f;

  for (int k0 = 0; k0 < Kd; k0 += BT_K) {
    // stage A tile 128x32: coalesced float4 loads, f16 LDS rows
#pragma unroll
    for (int i = 0; i < 4; ++i) {
      int idx4 = tid * 4 + i;          // 1024 float4
      int r = idx4 >> 3;               // 8 float4 per 32-wide row
      int k4 = (idx4 & 7) << 2;
      float4 a4 = *(const float4*)&A[(size_t)(rowBase + r) * Kd + k0 + k4];
      _Float16* dst = &As[r * LDS_S + k4];
      dst[0] = (_Float16)a4.x; dst[1] = (_Float16)a4.y;
      dst[2] = (_Float16)a4.z; dst[3] = (_Float16)a4.w;
    }
    // stage B tile 32x64 transposed: coalesced float4 along N, scatter to Bs[n][k]
#pragma unroll
    for (int i = 0; i < 2; ++i) {
      int idx4 = tid * 2 + i;          // 512 float4
      int k = idx4 >> 4;               // 16 float4 per 64-wide row
      int n4 = (idx4 & 15) << 2;
      float4 w4 = *(const float4*)&W[(size_t)(k0 + k) * N + colBase + n4];
      Bs[(n4 + 0) * LDS_S + k] = (_Float16)w4.x;
      Bs[(n4 + 1) * LDS_S + k] = (_Float16)w4.y;
      Bs[(n4 + 2) * LDS_S + k] = (_Float16)w4.z;
      Bs[(n4 + 3) * LDS_S + k] = (_Float16)w4.w;
    }
    if (k0 + BT_K < Kd)  // global_prefetch_b8 the next A tile
      __builtin_prefetch(&A[(size_t)(rowBase + (tid >> 1)) * Kd + k0 + BT_K], 0, 0);
    __syncthreads();

    v16h a0 = frag_lds16(As + (wr * 32) * LDS_S, LDS_S, lane);
    v16h a1 = frag_lds16(As + (wr * 32 + 16) * LDS_S, LDS_S, lane);
    v16h b0 = frag_lds16(Bs + (wc * 32) * LDS_S, LDS_S, lane);
    v16h b1 = frag_lds16(Bs + (wc * 32 + 16) * LDS_S, LDS_S, lane);
    acc[0][0] = WMMA_F16(a0, b0, acc[0][0]);
    acc[0][1] = WMMA_F16(a0, b1, acc[0][1]);
    acc[1][0] = WMMA_F16(a1, b0, acc[1][0]);
    acc[1][1] = WMMA_F16(a1, b1, acc[1][1]);
    __syncthreads();
  }

  int rb = (lane & 16) ? 8 : 0;
  int cl = lane & 15;
  (void)kb;
#pragma unroll
  for (int i = 0; i < 2; ++i)
#pragma unroll
    for (int j = 0; j < 2; ++j) {
      int col = colBase + wc * 32 + j * 16 + cl;
      float bv = bias ? bias[col] : 0.0f;
#pragma unroll
      for (int r = 0; r < 8; ++r) {
        int row = rowBase + wr * 32 + i * 16 + rb + r;
        float val = acc[i][j][r] + bv;
        if (resid) val += resid[(size_t)row * N + col];
        if (relu) val = fmaxf(val, 0.0f);
        C[(size_t)row * N + col] = val;
      }
    }
}

// ---------------- fused windowed attention (one 16-query block per WG, 4 waves) ----------------
// q: f16 [B*K][M][64]       (d contiguous)
// k: f16 [B*K][S+M][64]     (d contiguous)   -> QK^T B-frags = 2x b128 loads
// v_t: f16 [B*K][64][S+M]   (key contiguous) -> PV  B-frags = 2x b128 loads
// pe_t: f16 [S][64]         (d contiguous)   -> Q*PE B-frags = 2x b128 loads
#define ATT_LDS_BYTES (16 * 1072 * 4 + 128 * 4 + 16 * 72 * 2 + 16 * 1072 * 2)

__global__ __launch_bounds__(128) void attn_kernel(
    const _Float16* __restrict__ q, const _Float16* __restrict__ k,
    const _Float16* __restrict__ v_t, const _Float16* __restrict__ pe_t,
    float* __restrict__ outm) {
  extern __shared__ char smem[];
  float*    scores = (float*)smem;        // [16][1072]
  float*    red    = scores + 16 * 1072;  // [128]
  _Float16* qf     = (_Float16*)(red + 128);  // [16][72]
  _Float16* probs  = qf + 16 * 72;            // [16][1072]

  int i0 = blockIdx.x * 16;
  int bk = blockIdx.y;
  int b = bk / KH, hd = bk % KH;
  int tid = threadIdx.x, lane = tid & 31, wv = tid >> 5;
  int kb = (lane & 16) >> 1;
  int cl = lane & 15;
  int rb = (lane & 16) ? 8 : 0;

  const _Float16* Q  = q + ((size_t)bk * MM + i0) * DD;
  const _Float16* Km = k + (size_t)bk * (SS_ + MM) * DD;
  const _Float16* Vt = v_t + (size_t)bk * DD * (SS_ + MM);
  const int kkMax = SS_ + MM - 1;

  // phase 1: Q -> LDS (already f16); zero probability matrix (skew zero-pad)
  for (int idx = tid; idx < 16 * 64; idx += 128) {
    int r = idx >> 6, c = idx & 63;
    qf[r * 72 + c] = Q[r * 64 + c];
  }
  for (int idx = tid; idx < 16 * 1072; idx += 128) probs[idx] = (_Float16)0.0f;
  __syncthreads();

  // phase 2: content scores Q @ K_window^T  (66 column tiles of 16 keys)
  for (int ct = wv; ct < 66; ct += 4) {
    int kk = i0 + ct * 16 + cl;
    if (kk > kkMax) kk = kkMax;
    v8f acc = (v8f)0.0f;
#pragma unroll
    for (int ks = 0; ks < 64; ks += 32) {
      v16h a  = frag_lds16(qf + ks, 72, lane);
      v16h bb = frag_chunks(Km + (size_t)kk * DD + ks, kb);
      acc = WMMA_F16(a, bb, acc);
    }
#pragma unroll
    for (int r = 0; r < 8; ++r) scores[(rb + r) * 1072 + ct * 16 + cl] = acc[r];
  }
  __syncthreads();

  // phase 3: position scores Q @ PE, scattered with per-row shift (unskew fold-in)
  for (int ct = wv; ct < 64; ct += 4) {
    v8f acc = (v8f)0.0f;
#pragma unroll
    for (int ks = 0; ks < 64; ks += 32) {
      v16h a  = frag_lds16(qf + ks, 72, lane);
      v16h bb = frag_chunks(pe_t + (size_t)(ct * 16 + cl) * DD + ks, kb);
      acc = WMMA_F16(a, bb, acc);
    }
#pragma unroll
    for (int r = 0; r < 8; ++r) {
      int row = rb + r;
      scores[row * 1072 + (ct * 16 + cl) + row] += acc[r];  // jj = j + row
    }
  }
  __syncthreads();

  // phase 4: row-wise softmax over the 1024-wide window
  const float scale = 0.125f;  // 1/sqrt(64)
  for (int r = 0; r < 16; ++r) {
    float m = -3.4e38f;
    for (int jj = r + tid; jj < r + WIN; jj += 128) {
      float val = scores[r * 1072 + jj] * scale;
      scores[r * 1072 + jj] = val;
      m = fmaxf(m, val);
    }
    red[tid] = m; __syncthreads();
    for (int o = 64; o > 0; o >>= 1) { if (tid < o) red[tid] = fmaxf(red[tid], red[tid + o]); __syncthreads(); }
    m = red[0]; __syncthreads();
    float s = 0.0f;
    for (int jj = r + tid; jj < r + WIN; jj += 128) {
      float e = __expf(scores[r * 1072 + jj] - m);
      scores[r * 1072 + jj] = e;
      s += e;
    }
    red[tid] = s; __syncthreads();
    for (int o = 64; o > 0; o >>= 1) { if (tid < o) red[tid] += red[tid + o]; __syncthreads(); }
    float inv = 1.0f / red[0]; __syncthreads();
    for (int jj = r + tid; jj < r + WIN; jj += 128)
      probs[r * 1072 + jj] = (_Float16)(scores[r * 1072 + jj] * inv);
  }
  __syncthreads();

  // phase 5: out = P @ V_window; each wave owns a 16-col slice of D=64.
  // V^T rows are key-contiguous; chunk starts are multiples of 8, clamped so a
  // chunk never crosses the end of the sequence (clamped chunks only ever map
  // to zero-probability columns).
  {
    int n0 = wv * 16;
    const _Float16* vrow = Vt + (size_t)(n0 + cl) * (SS_ + MM);
    v8f acc = (v8f)0.0f;
    for (int kt = 0; kt < KEXT / 32; ++kt) {  // 33 K-steps of 32 keys
      int kBase = i0 + kt * 32;
      int s0 = kBase + kb;       if (s0 > (SS_ + MM - 8)) s0 = SS_ + MM - 8;
      int s1 = kBase + kb + 16;  if (s1 > (SS_ + MM - 8)) s1 = SS_ + MM - 8;
      v16h a  = frag_lds16(probs + kt * 32, 1072, lane);
      v16h bb = frag_chunks2(vrow, s0, s1);
      acc = WMMA_F16(a, bb, acc);
    }
#pragma unroll
    for (int r = 0; r < 8; ++r) {
      int row = i0 + rb + r;
      outm[((size_t)b * MM + row) * HH + hd * 64 + n0 + cl] = acc[r];  // merged heads
    }
  }
}

// ---------------- small helper kernels ----------------
__global__ __launch_bounds__(256) void embed_kernel(const int* __restrict__ x,
                                                    const float* __restrict__ emb,
                                                    float* __restrict__ h) {
  int row = blockIdx.x, t = threadIdx.x;
  const float* e = emb + (size_t)x[row] * HH;
  float* o = h + (size_t)row * HH;
  o[t] = e[t]; o[t + 256] = e[t + 256]; o[t + 512] = e[t + 512];
}

__global__ __launch_bounds__(256) void concat_kernel(const float* __restrict__ cache,
                                                     const float* __restrict__ h,
                                                     float* __restrict__ h_all) {
  int idx = blockIdx.x * 256 + threadIdx.x;  // B*1536*768 total
  int c = idx % HH;
  int t = (idx / HH) % (SS_ + MM);
  int b = idx / (HH * (SS_ + MM));
  float val;
  if (t < SS_) val = cache[((size_t)b * SS_ + t) * HH + c];
  else         val = h[((size_t)b * MM + (t - SS_)) * HH + c];
  h_all[idx] = val;
}

// [B,T,H] fp32 -> f16 head-split [B*K][T][64] (d contiguous)
__global__ __launch_bounds__(256) void split_heads_f16_kernel(const float* __restrict__ in,
                                                              _Float16* __restrict__ out, int T) {
  int idx = blockIdx.x * 256 + threadIdx.x;
  int c = idx % HH;
  int t = (idx / HH) % T;
  int b = idx / (HH * T);
  int hd = c >> 6, d = c & 63;
  out[(((size_t)b * KH + hd) * T + t) * DD + d] = (_Float16)in[idx];
}

// [B,T,H] fp32 -> f16 transposed per head [B*K][64][T] (key contiguous)
__global__ __launch_bounds__(256) void transpose_v_f16_kernel(const float* __restrict__ in,
                                                              _Float16* __restrict__ out, int T) {
  int idx = blockIdx.x * 256 + threadIdx.x;
  int c = idx % HH;
  int t = (idx / HH) % T;
  int b = idx / (HH * T);
  int hd = c >> 6, d = c & 63;
  out[(((size_t)b * KH + hd) * DD + d) * T + t] = (_Float16)in[idx];
}

// key_pe [64][1024] fp32 -> pe_t [1024][64] f16 (d contiguous)
__global__ __launch_bounds__(256) void pe_transpose_kernel(const float* __restrict__ pe,
                                                           _Float16* __restrict__ pe_t) {
  int idx = blockIdx.x * 256 + threadIdx.x;  // 65536
  int j = idx >> 6, d = idx & 63;
  pe_t[idx] = (_Float16)pe[d * SS_ + j];
}

__global__ __launch_bounds__(256) void ln_kernel(const float* __restrict__ in,
                                                 const float* __restrict__ sc,
                                                 const float* __restrict__ bi,
                                                 float* __restrict__ out) {
  __shared__ float red[256];
  int row = blockIdx.x, t = threadIdx.x;
  const float* x = in + (size_t)row * HH;
  float a0 = x[t], a1 = x[t + 256], a2 = x[t + 512];
  red[t] = a0 + a1 + a2; __syncthreads();
  for (int o = 128; o > 0; o >>= 1) { if (t < o) red[t] += red[t + o]; __syncthreads(); }
  float mu = red[0] * (1.0f / HH); __syncthreads();
  float d0 = a0 - mu, d1 = a1 - mu, d2 = a2 - mu;
  red[t] = d0 * d0 + d1 * d1 + d2 * d2; __syncthreads();
  for (int o = 128; o > 0; o >>= 1) { if (t < o) red[t] += red[t + o]; __syncthreads(); }
  float rinv = rsqrtf(red[0] * (1.0f / HH) + 1e-5f);
  float* o = out + (size_t)row * HH;
  o[t]       = d0 * rinv * sc[t]       + bi[t];
  o[t + 256] = d1 * rinv * sc[t + 256] + bi[t + 256];
  o[t + 512] = d2 * rinv * sc[t + 512] + bi[t + 512];
}

__global__ __launch_bounds__(256) void log_softmax_kernel(float* __restrict__ io) {
  __shared__ float red[256];
  int row = blockIdx.x, t = threadIdx.x;
  float* x = io + (size_t)row * VV;
  float m = -3.4e38f;
  for (int j = t; j < VV; j += 256) m = fmaxf(m, x[j]);
  red[t] = m; __syncthreads();
  for (int o = 128; o > 0; o >>= 1) { if (t < o) red[t] = fmaxf(red[t], red[t + o]); __syncthreads(); }
  m = red[0]; __syncthreads();
  float s = 0.0f;
  for (int j = t; j < VV; j += 256) s += __expf(x[j] - m);
  red[t] = s; __syncthreads();
  for (int o = 128; o > 0; o >>= 1) { if (t < o) red[t] += red[t + o]; __syncthreads(); }
  float lse = __logf(red[0]) + m;
  for (int j = t; j < VV; j += 256) x[j] = x[j] - lse;
}

// ---------------- host orchestration ----------------
extern "C" void kernel_launch(void* const* d_in, const int* in_sizes, int n_in,
                              void* d_out, int out_size, void* d_ws, size_t ws_size,
                              hipStream_t stream) {
  (void)in_sizes; (void)n_in; (void)out_size; (void)ws_size;
  const int*   x       = (const int*)d_in[0];
  const float* h_cache = (const float*)d_in[1];
  const float* key_pe  = (const float*)d_in[2];
  const float* emb     = (const float*)d_in[3];
  const float* Wq      = (const float*)d_in[4];
  const float* Wk      = (const float*)d_in[5];
  const float* Wv      = (const float*)d_in[6];
  const float* Wo      = (const float*)d_in[7];
  const float* fc1_w   = (const float*)d_in[8];
  const float* fc1_b   = (const float*)d_in[9];
  const float* fc2_w   = (const float*)d_in[10];
  const float* fc2_b   = (const float*)d_in[11];
  const float* ln1_s   = (const float*)d_in[12];
  const float* ln1_b   = (const float*)d_in[13];
  const float* ln2_s   = (const float*)d_in[14];
  const float* ln2_b   = (const float*)d_in[15];
  const float* out_w   = (const float*)d_in[16];
  const float* out_b   = (const float*)d_in[17];
  float* out = (float*)d_out;

  // workspace partitions (256B-aligned byte offsets)
  char* base = (char*)d_ws;
  size_t off = 0;
  auto align_up = [](size_t v) { return (v + 255) & ~(size_t)255; };
  float*    h_buf  = (float*)(base + off);    off += align_up((size_t)TQ * HH * 4);
  float*    tmp    = (float*)(base + off);    off += align_up((size_t)TQ * HH * 4);
  float*    h_all  = (float*)(base + off);    off += align_up((size_t)TKV * HH * 4);
  float*    qkv    = (float*)(base + off);    off += align_up((size_t)TKV * HH * 4);
  _Float16* q_h    = (_Float16*)(base + off); off += align_up((size_t)BB * KH * MM * DD * 2);
  _Float16* k_h    = (_Float16*)(base + off); off += align_up((size_t)BB * KH * (SS_ + MM) * DD * 2);
  _Float16* v_t    = (_Float16*)(base + off); off += align_up((size_t)BB * KH * DD * (SS_ + MM) * 2 + 256);
  _Float16* pe_t   = (_Float16*)(base + off); off += align_up((size_t)DD * SS_ * 2);
  float*    attn_m = (float*)(base + off);    off += align_up((size_t)TQ * HH * 4);
  float*    ff     = (float*)(base + off);    off += align_up((size_t)TQ * FF_ * 4);

  (void)hipFuncSetAttribute((const void*)attn_kernel,
                            hipFuncAttributeMaxDynamicSharedMemorySize, ATT_LDS_BYTES);

  embed_kernel<<<TQ, 256, 0, stream>>>(x, emb, h_buf);
  pe_transpose_kernel<<<(DD * SS_) / 256, 256, 0, stream>>>(key_pe, pe_t);

  for (int l = 0; l < NL_; ++l) {
    const float* cache_l = h_cache + (size_t)l * BB * SS_ * HH;
    concat_kernel<<<(TKV * HH) / 256, 256, 0, stream>>>(cache_l, h_buf, h_all);

    // Q/K/V projections (WMMA GEMM) + f16 head split / V transpose
    gemm_wmma_kernel<<<dim3(HH / BT_N, TQ / BT_M), 256, 0, stream>>>(
        h_buf, Wq + (size_t)l * HH * HH, nullptr, nullptr, qkv, TQ, HH, HH, 0);
    split_heads_f16_kernel<<<(TQ * HH) / 256, 256, 0, stream>>>(qkv, q_h, MM);

    gemm_wmma_kernel<<<dim3(HH / BT_N, TKV / BT_M), 256, 0, stream>>>(
        h_all, Wk + (size_t)l * HH * HH, nullptr, nullptr, qkv, TKV, HH, HH, 0);
    split_heads_f16_kernel<<<(TKV * HH) / 256, 256, 0, stream>>>(qkv, k_h, SS_ + MM);

    gemm_wmma_kernel<<<dim3(HH / BT_N, TKV / BT_M), 256, 0, stream>>>(
        h_all, Wv + (size_t)l * HH * HH, nullptr, nullptr, qkv, TKV, HH, HH, 0);
    transpose_v_f16_kernel<<<(TKV * HH) / 256, 256, 0, stream>>>(qkv, v_t, SS_ + MM);

    // fused windowed attention -> merged heads [B,M,H]
    attn_kernel<<<dim3(MM / 16, BB * KH), 128, ATT_LDS_BYTES, stream>>>(
        q_h, k_h, v_t, pe_t, attn_m);

    // output projection + residual, then LN1
    gemm_wmma_kernel<<<dim3(HH / BT_N, TQ / BT_M), 256, 0, stream>>>(
        attn_m, Wo + (size_t)l * HH * HH, nullptr, h_buf, tmp, TQ, HH, HH, 0);
    ln_kernel<<<TQ, 256, 0, stream>>>(tmp, ln1_s + l * HH, ln1_b + l * HH, h_buf);

    // FFN: relu(h@W1+b1)@W2+b2 + residual, then LN2
    gemm_wmma_kernel<<<dim3(FF_ / BT_N, TQ / BT_M), 256, 0, stream>>>(
        h_buf, fc1_w + (size_t)l * HH * FF_, fc1_b + (size_t)l * FF_, nullptr, ff, TQ, FF_, HH, 1);
    gemm_wmma_kernel<<<dim3(HH / BT_N, TQ / BT_M), 256, 0, stream>>>(
        ff, fc2_w + (size_t)l * FF_ * HH, fc2_b + (size_t)l * HH, h_buf, tmp, TQ, HH, FF_, 0);
    ln_kernel<<<TQ, 256, 0, stream>>>(tmp, ln2_s + l * HH, ln2_b + l * HH, h_buf);
  }

  // vocab projection + log_softmax (writes d_out fully)
  gemm_wmma_kernel<<<dim3(VV / BT_N, TQ / BT_M), 256, 0, stream>>>(
      h_buf, out_w, out_b, nullptr, out, TQ, VV, HH, 0);
  log_softmax_kernel<<<TQ, 256, 0, stream>>>(out);
}
